// AutoRegressiveDecoderLayer_9414568313545
// MI455X (gfx1250) — matrix operand order repaired and checked by reading, not compiled, not run on hardware
//
#include <hip/hip_runtime.h>
#include <hip/hip_bf16.h>
#include <stddef.h>

#define DIM 128
#define HEADS 8
#define DH 16
#define BSZ 1024
#define NODES 1000
#define T_PREV 500

typedef __attribute__((ext_vector_type(16))) __bf16 v16bf;
typedef __attribute__((ext_vector_type(8)))  float  v8f;
typedef __attribute__((ext_vector_type(8)))  unsigned v8u;

// pack two f32 into one dword of 2 bf16 (truncation) with a single v_perm_b32
__device__ __forceinline__ unsigned pack_bf2(float lo, float hi) {
    return __builtin_amdgcn_perm(__builtin_bit_cast(unsigned, hi),
                                 __builtin_bit_cast(unsigned, lo),
                                 0x07060302u);
}

// A-fragment element j of a lane -> k offset within the 32-wide chunk
// (ISA 7.12.2, 16-bit A 16x32: VGPR0..3 = K octet 0/1 per half, VGPR4..7 = +16)
__device__ __forceinline__ int a_koff(int j, int half) {
    return ((j >> 3) << 4) + (half << 3) + (j & 7);
}

// LDS byte offset of a __shared__ pointer (flat->local lowers to truncation)
__device__ __forceinline__ unsigned lds_off(const void* p) {
    return (unsigned)(unsigned long long)p;
}

// A fragment from pre-packed bf16-pair LDS tile (row mn, chunk k0)
__device__ __forceinline__ v16bf a_from_pairs(const unsigned* xp, int k0, int half) {
    v8u au;
#pragma unroll
    for (int jj = 0; jj < 8; ++jj) {
        const int p = (k0 >> 1) + (half << 2) + (jj & 3) + ((jj >> 2) << 3);
        au[jj] = xp[p];
    }
    return __builtin_bit_cast(v16bf, au);
}

// B fragment (B[k][n] = W[n][k]) for y = x @ W^T, packed from global f32 W row
__device__ __forceinline__ v16bf b_from_w(const float* __restrict__ wrow, int k0, int half) {
    v8u bu;
#pragma unroll
    for (int jj = 0; jj < 8; ++jj) {
        const int k = k0 + ((jj >> 2) << 4) + (half << 3) + 2 * (jj & 3);
        bu[jj] = pack_bf2(wrow[k], wrow[k + 1]);
    }
    return __builtin_bit_cast(v16bf, bu);
}

// ---------------------------------------------------------------------------
// y[b0..b0+15][0..127] = act(x @ W^T + bias); W is (N=128, K=128) row-major
// ---------------------------------------------------------------------------
__global__ __launch_bounds__(256) void gemm_bias_kernel(
    const float* __restrict__ x, const float* __restrict__ W,
    const float* __restrict__ bias, float* __restrict__ y, int relu)
{
    __shared__ unsigned xsp[16][DIM / 2];   // bf16-pair packed x tile
    const int b0 = blockIdx.x * 16;
    const int tid = threadIdx.x;

    for (int i = tid; i < 16 * (DIM / 2); i += 256) {
        const int r = i >> 6, c2 = i & 63;
        const float* xp = x + (size_t)(b0 + r) * DIM + c2 * 2;
        xsp[r][c2] = pack_bf2(xp[0], xp[1]);
    }
    __syncthreads();

    const int wave = tid >> 5, lane = tid & 31;
    const int half = lane >> 4, mn = lane & 15;
    const int n0 = wave * 16;
    const float* wrow = W + (size_t)(n0 + mn) * DIM;

    v8f acc = {};
#pragma unroll
    for (int kc = 0; kc < 4; ++kc) {
        const v16bf a  = a_from_pairs(xsp[mn], kc * 32, half);
        const v16bf bm = b_from_w(wrow, kc * 32, half);
        acc = __builtin_amdgcn_wmma_f32_16x16x32_bf16(false, a, false, bm,
                                                      (short)0, acc, false, false);
    }

    const float bv = bias[n0 + mn];
#pragma unroll
    for (int r = 0; r < 8; ++r) {
        float v = acc[r] + bv;
        if (relu) v = fmaxf(v, 0.0f);
        const int row = (half << 3) + r;
        y[(size_t)(b0 + row) * DIM + n0 + mn] = v;
    }
}

// ---------------------------------------------------------------------------
// out = LayerNorm(resid + x @ W^T + bias) * gamma + beta
// ---------------------------------------------------------------------------
__global__ __launch_bounds__(256) void gemm_res_ln_kernel(
    const float* __restrict__ x, const float* __restrict__ W,
    const float* __restrict__ bias, const float* __restrict__ resid,
    const float* __restrict__ gamma, const float* __restrict__ beta,
    float* __restrict__ out)
{
    __shared__ unsigned xsp[16][DIM / 2];
    __shared__ float  ys[16][DIM];
    __shared__ float  mu[16], rs[16];
    const int b0 = blockIdx.x * 16;
    const int tid = threadIdx.x;

    for (int i = tid; i < 16 * (DIM / 2); i += 256) {
        const int r = i >> 6, c2 = i & 63;
        const float* xp = x + (size_t)(b0 + r) * DIM + c2 * 2;
        xsp[r][c2] = pack_bf2(xp[0], xp[1]);
    }
    __syncthreads();

    const int wave = tid >> 5, lane = tid & 31;
    const int half = lane >> 4, mn = lane & 15;
    const int n0 = wave * 16;
    const float* wrow = W + (size_t)(n0 + mn) * DIM;

    v8f acc = {};
#pragma unroll
    for (int kc = 0; kc < 4; ++kc) {
        const v16bf a  = a_from_pairs(xsp[mn], kc * 32, half);
        const v16bf bm = b_from_w(wrow, kc * 32, half);
        acc = __builtin_amdgcn_wmma_f32_16x16x32_bf16(false, a, false, bm,
                                                      (short)0, acc, false, false);
    }

    const float bv = bias[n0 + mn];
#pragma unroll
    for (int r = 0; r < 8; ++r) {
        const int row = (half << 3) + r;
        ys[row][n0 + mn] = acc[r] + bv + resid[(size_t)(b0 + row) * DIM + n0 + mn];
    }
    __syncthreads();

    if (tid < 16) {
        float m = 0.f;
        for (int c = 0; c < DIM; ++c) m += ys[tid][c];
        m *= (1.0f / DIM);
        float v = 0.f;
        for (int c = 0; c < DIM; ++c) { float d = ys[tid][c] - m; v += d * d; }
        v *= (1.0f / DIM);
        mu[tid] = m;
        rs[tid] = rsqrtf(v + 1e-5f);
    }
    __syncthreads();

    for (int i = tid; i < 16 * DIM; i += 256) {
        const int r = i >> 7, c = i & (DIM - 1);
        out[(size_t)(b0 + r) * DIM + c] = (ys[r][c] - mu[r]) * rs[r] * gamma[c] + beta[c];
    }
}

// ---------------------------------------------------------------------------
// Async-stage one 16x128 f32 K tile into LDS: 16 rows, one
// global_load_async_to_lds_b128 per row (16B per lane, fully coalesced).
// ---------------------------------------------------------------------------
__device__ __forceinline__ void issue_tile_async(
    float (*dst)[DIM], const float* __restrict__ Kb, const float* __restrict__ Kx,
    int b, int t, int n_main, int n_total, int lane)
{
#pragma unroll
    for (int i = 0; i < 16; ++i) {
        const int node = min(t * 16 + i, n_total - 1);
        const float* rp = (node < n_main) ? (Kb + (size_t)node * DIM)
                                          : (Kx + (size_t)b * DIM);
        const unsigned loff = lds_off(&dst[i][0]) + lane * 16;
        const void* ga = (const void*)(rp + lane * 4);
        asm volatile("global_load_async_to_lds_b128 %0, %1, off"
                     :: "v"(loff), "v"(ga) : "memory");
    }
}

// ---------------------------------------------------------------------------
// Multi-head attention, one query per batch element, 8 heads x dh=16.
// Scores: WMMA of K tile (16 nodes x 128) vs block-diagonal q.
// K tiles are double-buffered in LDS via async loads (ASYNCcnt).
// ---------------------------------------------------------------------------
__global__ __launch_bounds__(256) void attn_kernel(
    const float* __restrict__ Q,     // (B,128)
    const float* __restrict__ Kmat,  // (B, n_main, 128)
    const float* __restrict__ Vmat,  // (B, n_main, 128)
    const float* __restrict__ Kx,    // (B,128) appended row or nullptr
    const float* __restrict__ Vx,
    int n_main, int n_total,
    const unsigned char* __restrict__ mask,  // (B, n_total) bool or nullptr
    float* __restrict__ O)           // (B,128)
{
    const int b    = blockIdx.x;
    const int tid  = threadIdx.x;
    const int wave = tid >> 5, lane = tid & 31;
    const int half = lane >> 4, hn = lane & 15;

    __shared__ float kstage[8][2][16][DIM];         // 128 KB: per-wave double buffer
    __shared__ float qf[DIM];
    __shared__ float red[8][HEADS][2 + DH];         // [wave][head][m, s, acc0..15]

    if (tid < DIM) qf[tid] = Q[(size_t)b * DIM + tid];
    __syncthreads();

    // Block-diagonal B fragments: column h nonzero only for dims [16h, 16h+16)
    v16bf bfrag[4];
#pragma unroll
    for (int kc = 0; kc < 4; ++kc) {
        v8u bu;
#pragma unroll
        for (int jj = 0; jj < 8; ++jj) {
            const int kg = kc * 32 + ((jj >> 2) << 4) + (half << 3) + 2 * (jj & 3);
            bu[jj] = (hn == (kg >> 4)) ? pack_bf2(qf[kg], qf[kg + 1]) : 0u;
        }
        bfrag[kc] = __builtin_bit_cast(v16bf, bu);
    }

    const float* Kb = Kmat + (size_t)b * n_main * DIM;
    const float* Vb = Vmat + (size_t)b * n_main * DIM;

    float m = -1e30f, s = 0.0f;
    float acc[DH];
#pragma unroll
    for (int d = 0; d < DH; ++d) acc[d] = 0.0f;

    const int ntiles = (n_total + 15) >> 4;

    if (wave < ntiles)
        issue_tile_async(kstage[wave][0], Kb, Kx, b, wave, n_main, n_total, lane);

    int bufsel = 0;
    for (int t = wave; t < ntiles; t += 8, bufsel ^= 1) {
        const int tn = t + 8;
        const bool more = (tn < ntiles);
        if (more) {
            issue_tile_async(kstage[wave][bufsel ^ 1], Kb, Kx, b, tn,
                             n_main, n_total, lane);
            // async loads complete in order: <=16 outstanding => tile t resident
            asm volatile("s_wait_asynccnt 0x10" ::: "memory");
        } else {
            asm volatile("s_wait_asynccnt 0x0" ::: "memory");
        }
        const float (*ks)[DIM] = kstage[wave][bufsel];
        const int node0 = t * 16;

        // prefetch next tile's V rows for the active (head) lanes
        if (hn < HEADS) {
            const int nn = node0 + 128 + (half << 3);
            if (nn < n_main) __builtin_prefetch(Vb + (size_t)nn * DIM + hn * DH, 0, 0);
        }

        // --- scores: 4 WMMA over K=128, A rows from the staged LDS tile ---
        v8f sc = {};
#pragma unroll
        for (int kc = 0; kc < 4; ++kc) {
            v8u au;
#pragma unroll
            for (int jj = 0; jj < 8; ++jj) {
                const int k = kc * 32 + a_koff(2 * jj, half);
                au[jj] = pack_bf2(ks[hn][k], ks[hn][k + 1]);
            }
            const v16bf a = __builtin_bit_cast(v16bf, au);
            sc = __builtin_amdgcn_wmma_f32_16x16x32_bf16(false, a, false, bfrag[kc],
                                                         (short)0, sc, false, false);
        }

        // --- per-lane online softmax (head = hn, rows node0 + 8*half + r) ---
        float scr[8], tm = -1e30f;
#pragma unroll
        for (int r = 0; r < 8; ++r) {
            const int node = node0 + (half << 3) + r;
            float v = sc[r] * 0.25f;  // 1/sqrt(dh), dh=16
            bool bad = (node >= n_total);
            if (!bad && mask) bad = (mask[(size_t)b * n_total + node] != 0);
            scr[r] = bad ? -1e9f : v;
            tm = fmaxf(tm, scr[r]);
        }
        const float nm = fmaxf(m, tm);
        const float fold = __expf(m - nm);
        s *= fold;
#pragma unroll
        for (int d = 0; d < DH; ++d) acc[d] *= fold;

#pragma unroll
        for (int r = 0; r < 8; ++r) {
            const float p = __expf(scr[r] - nm);
            s += p;
            if (hn < HEADS) {
                const int node = min(node0 + (half << 3) + r, n_total - 1);
                const float* vr = (node < n_main) ? (Vb + (size_t)node * DIM)
                                                  : (Vx + (size_t)b * DIM);
#pragma unroll
                for (int d = 0; d < DH; ++d)
                    acc[d] = fmaf(p, vr[hn * DH + d], acc[d]);
            }
        }
        m = nm;
    }

    // --- merge the two lanes of each head within the wave (lane ^ 16) ---
    const float m2 = __shfl_xor(m, 16, 32);
    const float s2 = __shfl_xor(s, 16, 32);
    const float M  = fmaxf(m, m2);
    const float f1 = __expf(m - M), f2x = __expf(m2 - M);
    const float S  = s * f1 + s2 * f2x;
    float accm[DH];
#pragma unroll
    for (int d = 0; d < DH; ++d) {
        const float a2 = __shfl_xor(acc[d], 16, 32);
        accm[d] = acc[d] * f1 + a2 * f2x;
    }
    if (lane < HEADS) {
        red[wave][lane][0] = M;
        red[wave][lane][1] = S;
#pragma unroll
        for (int d = 0; d < DH; ++d) red[wave][lane][2 + d] = accm[d];
    }
    __syncthreads();

    // --- final merge across 8 waves: thread = head*16 + dim ---
    if (tid < HEADS * DH) {
        const int h = tid >> 4, d = tid & 15;
        float Mg = -1e30f;
        for (int w = 0; w < 8; ++w) Mg = fmaxf(Mg, red[w][h][0]);
        float Sg = 0.f, Ag = 0.f;
        for (int w = 0; w < 8; ++w) {
            const float f = __expf(red[w][h][0] - Mg);
            Sg += red[w][h][1] * f;
            Ag += red[w][h][2 + d] * f;
        }
        O[(size_t)b * DIM + h * DH + d] = Ag / Sg;
    }
}

// ---------------------------------------------------------------------------
extern "C" void kernel_launch(void* const* d_in, const int* in_sizes, int n_in,
                              void* d_out, int out_size, void* d_ws, size_t ws_size,
                              hipStream_t stream) {
    const float* h_t       = (const float*)d_in[0];
    const float* K_att     = (const float*)d_in[1];
    const float* V_att     = (const float*)d_in[2];
    const float* K_sa_prev = (const float*)d_in[3];
    const float* V_sa_prev = (const float*)d_in[4];
    const float* Wq_sa = (const float*)d_in[5];  const float* bq_sa = (const float*)d_in[6];
    const float* Wk_sa = (const float*)d_in[7];  const float* bk_sa = (const float*)d_in[8];
    const float* Wv_sa = (const float*)d_in[9];  const float* bv_sa = (const float*)d_in[10];
    const float* W0_sa = (const float*)d_in[11]; const float* b0_sa = (const float*)d_in[12];
    const float* Wq_a  = (const float*)d_in[13]; const float* bq_a  = (const float*)d_in[14];
    const float* W0_a  = (const float*)d_in[15]; const float* b0_a  = (const float*)d_in[16];
    const float* W1    = (const float*)d_in[17]; const float* b1    = (const float*)d_in[18];
    const float* W2    = (const float*)d_in[19]; const float* b2    = (const float*)d_in[20];
    const float* g_sa  = (const float*)d_in[21]; const float* be_sa = (const float*)d_in[22];
    const float* g_a   = (const float*)d_in[23]; const float* be_a  = (const float*)d_in[24];
    const float* g_mlp = (const float*)d_in[25]; const float* be_mlp= (const float*)d_in[26];
    const unsigned char* mask = (const unsigned char*)d_in[27];

    const size_t N = (size_t)BSZ * DIM;
    float* ws   = (float*)d_ws;
    float* q_sa = ws + 0 * N;
    float* k_sa = ws + 1 * N;
    float* v_sa = ws + 2 * N;
    float* o_sa = ws + 3 * N;
    float* h1   = ws + 4 * N;
    float* q_a  = ws + 5 * N;
    float* o_a  = ws + 6 * N;
    float* h2   = ws + 7 * N;
    float* mlp  = ws + 8 * N;
    float* outp = (float*)d_out;

    const dim3 gG(BSZ / 16), gA(BSZ), blk(256);

    // q/k/v projections
    gemm_bias_kernel<<<gG, blk, 0, stream>>>(h_t, Wq_sa, bq_sa, q_sa, 0);
    gemm_bias_kernel<<<gG, blk, 0, stream>>>(h_t, Wk_sa, bk_sa, k_sa, 0);
    gemm_bias_kernel<<<gG, blk, 0, stream>>>(h_t, Wv_sa, bv_sa, v_sa, 0);
    // self-attention over cache (500) + appended row (1)
    attn_kernel<<<gA, blk, 0, stream>>>(q_sa, K_sa_prev, V_sa_prev, k_sa, v_sa,
                                        T_PREV, T_PREV + 1, nullptr, o_sa);
    gemm_res_ln_kernel<<<gG, blk, 0, stream>>>(o_sa, W0_sa, b0_sa, h_t, g_sa, be_sa, h1);
    // masked cross-attention over node embeddings
    gemm_bias_kernel<<<gG, blk, 0, stream>>>(h1, Wq_a, bq_a, q_a, 0);
    attn_kernel<<<gA, blk, 0, stream>>>(q_a, K_att, V_att, nullptr, nullptr,
                                        NODES, NODES, mask, o_a);
    gemm_res_ln_kernel<<<gG, blk, 0, stream>>>(o_a, W0_a, b0_a, h1, g_a, be_a, h2);
    // MLP
    gemm_bias_kernel<<<gG, blk, 0, stream>>>(h2, W1, b1, mlp, 1);
    gemm_res_ln_kernel<<<gG, blk, 0, stream>>>(mlp, W2, b2, h2, g_mlp, be_mlp, outp);
}
